// PredictionEncoder_86079734546935
// MI455X (gfx1250) — matrix operand order repaired and checked by reading, not compiled, not run on hardware
//
#include <hip/hip_runtime.h>

// ---------------------------------------------------------------------------
// CDNA5 (gfx1250) implementation. wave32. Heavy GEMMs: v_wmma_f32_16x16x32_bf16
// with 4 independent accumulator chains per wave (16x64 tile), LDS-staged A
// operand (async-to-LDS), TDM-staged B operand for the last GEMM.
// ---------------------------------------------------------------------------

typedef __attribute__((ext_vector_type(16))) __bf16       v16bf;
typedef __attribute__((ext_vector_type(8)))  __bf16       v8bf;
typedef __attribute__((ext_vector_type(8)))  float        v8f;
typedef __attribute__((ext_vector_type(4)))  unsigned int u32x4;
typedef __attribute__((ext_vector_type(8)))  int          i32x8;
typedef __attribute__((ext_vector_type(4)))  int          i32x4;

#define AS1 __attribute__((address_space(1)))
#define AS3 __attribute__((address_space(3)))

#if __has_builtin(__builtin_amdgcn_global_load_async_to_lds_b128) && \
    __has_builtin(__builtin_amdgcn_s_wait_asynccnt)
#define USE_ASYNC_LDS 1
#endif
#if __has_builtin(__builtin_amdgcn_tensor_load_to_lds) && \
    __has_builtin(__builtin_amdgcn_s_wait_tensorcnt)
#define USE_TDM 1
#endif

#define Nn    256
#define Vv    256
#define Hh    256
#define LAT   64
#define OUTC  128
#define Mrows (Nn * Vv)          // 65536
#define GIN   (2 * Hh + LAT)     // 576
#define CF_K  (2 * Hh)           // 512

static __device__ __forceinline__ __bf16 f2bf(float f) { return (__bf16)f; }

static __device__ __forceinline__ v8f wmma_bf16(const v16bf& a, const v16bf& b, v8f c) {
  return __builtin_amdgcn_wmma_f32_16x16x32_bf16(false, a, false, b, (short)0, c,
                                                 false, false);
}

// ---------------------------------------------------------------------------
// Weight prep: f32 (K x C) row-major -> bf16 transposed (C x K) row-major.
// ---------------------------------------------------------------------------
__global__ void wprep_kernel(const float* __restrict__ W, __bf16* __restrict__ Wt,
                             int K, int C) {
  long idx = (long)blockIdx.x * 256 + threadIdx.x;
  if (idx >= (long)K * C) return;
  int  c = (int)(idx % C);
  long k = idx / C;
  Wt[(long)c * K + k] = f2bf(W[idx]);
}

// ---------------------------------------------------------------------------
// Embed: se/ve + comb assembly; se also lands in cf[:,256:512].
// ---------------------------------------------------------------------------
__global__ void embed_kernel(const float* __restrict__ obs, const float* __restrict__ hi,
                             const float* __restrict__ Ws, const float* __restrict__ bs,
                             const float* __restrict__ Wv, const float* __restrict__ bv,
                             __bf16* __restrict__ comb, __bf16* __restrict__ cf) {
  long m = blockIdx.x;
  int  t = threadIdx.x;
  const float* o = obs + m * 5;
  float o1 = o[1], o2 = o[2], o3 = o[3], o4 = o[4];
  float se = o1 * Ws[t] + o2 * Ws[Hh + t] + bs[t];
  se = se > 0.f ? se : 0.f;
  float ve = o3 * Wv[t] + o4 * Wv[Hh + t] + bv[t];
  ve = ve > 0.f ? ve : 0.f;
  __bf16* cr = comb + m * GIN;
  cr[t]      = f2bf(se);
  cr[Hh + t] = f2bf(ve);
  if (t < LAT) cr[2 * Hh + t] = f2bf(hi[m * LAT + t]);
  cf[m * CF_K + Hh + t] = f2bf(se);
}

// ---------------------------------------------------------------------------
// bf16 WMMA GEMM:  C[M,Nc] = A[M,K] @ Bt[Nc,K]^T (+bias).
// Block = 8 waves; wave computes 16x64 (4 independent WMMA chains).
// A rows for the block staged in LDS (padded stride, async-to-LDS).
// TDMB: whole Bt (Nc=128,K=256, 64KB) staged via TENSOR_LOAD_TO_LDS with
//       LDS padding (row stride 264 elems) for conflict-free B fragments.
// ---------------------------------------------------------------------------
template <bool HASBIAS, bool TDMB>
__global__ void gemm_bf16_kernel(const __bf16* __restrict__ A,
                                 const __bf16* __restrict__ Bt,
                                 const float* __restrict__ bias,
                                 float* __restrict__ Cf,
                                 int K, int Nc) {
  extern __shared__ char smem[];
  const int lane = threadIdx.x & 31;
  const int wave = threadIdx.x >> 5;
  const int wavesPerRT = Nc >> 6;          // waves covering one 16-row tile
  const int RB = (8 / wavesPerRT) << 4;    // rows of A per block (32 or 64)
  const int sA = K + 8;                    // padded LDS stride (elems)
  __bf16* ldsA = (__bf16*)smem;
  __bf16* ldsB = (__bf16*)(smem + (size_t)RB * sA * 2);

  const long m0b = (long)blockIdx.x * RB;

  // ---- stage A tile rows into LDS ----
  {
    const int cpr = K >> 3;                // 16B chunks per row
    const int total = RB * cpr;
    for (int c = threadIdx.x; c < total; c += 256) {
      int row = c / cpr, c8 = c - row * cpr;
      const __bf16* g = A + (m0b + row) * K + (c8 << 3);
      __bf16*       l = ldsA + row * sA + (c8 << 3);
#ifdef USE_ASYNC_LDS
      __builtin_amdgcn_global_load_async_to_lds_b128((AS1 i32x4*)g, (AS3 i32x4*)l, 0, 0);
#else
      *(v8bf*)l = *(const v8bf*)g;
#endif
    }
  }

  // ---- stage B via Tensor Data Mover (one descriptor, whole weight) ----
  if constexpr (TDMB) {
#ifdef USE_TDM
    if (wave == 0) {
      unsigned ldsOff = (unsigned)(unsigned long long)(AS3 char*)ldsB;
      unsigned long long ga = (unsigned long long)Bt;
      // D# group0: count=1, lds_addr, global_addr[56:0], type=2
      u32x4 g0 = { 1u, ldsOff, (unsigned)ga,
                   (unsigned)((ga >> 32) & 0x01FFFFFFull) | (2u << 30) };
      // D# group1: data_size=2B, pad_enable, pad_interval=128dw, pad_amount=4dw,
      // tensor_dim0=256, tensor_dim1=128, tile_dim0=256, tile_dim1=128,
      // tensor_dim0_stride=256
      i32x8 g1 = { 0x07910000,          // mask=0|ds=1|pad_en|int=6|amt=3
                   (int)(256u << 16),   // tensor_dim0 lo16 @ [63:48]
                   (int)(128u << 16),   // tensor_dim1 lo16 @ [95:80]
                   (int)(256u << 16),   // tile_dim0 @ [127:112]
                   128,                 // tile_dim1 @ [143:128]
                   256,                 // tensor_dim0_stride lo32
                   0, 0 };
      i32x4 z4 = { 0, 0, 0, 0 };
      i32x8 z8 = { 0, 0, 0, 0, 0, 0, 0, 0 };
      __builtin_amdgcn_tensor_load_to_lds(g0, g1, z4, z4, z8, 0);
      __builtin_amdgcn_s_wait_tensorcnt(0);
    }
#else
    for (int c = threadIdx.x; c < 128 * 32; c += 256) {   // 64KB in 16B chunks
      int row = c >> 5, c8 = c & 31;
      *(v8bf*)(ldsB + row * 264 + (c8 << 3)) =
          *(const v8bf*)(Bt + row * 256 + (c8 << 3));
    }
#endif
  }
#ifdef USE_ASYNC_LDS
  __builtin_amdgcn_s_wait_asynccnt(0);
#endif
  __syncthreads();

  // ---- compute ----
  const int rt = wave / wavesPerRT;
  const int n0 = (wave % wavesPerRT) << 6;
  const int kSelA = (lane >> 4) << 4;      // 0 or 16
  const int kSelB = (lane >> 4) << 4;      // 0 or 16
  const __bf16* aLds = ldsA + (rt * 16 + (lane & 15)) * sA + kSelA;

  const long bStride = TDMB ? 264 : K;
  const __bf16* bBase = TDMB ? ldsB : Bt;
  const __bf16* bp0 = bBase + (long)(n0 +  0 + (lane & 15)) * bStride + kSelB;
  const __bf16* bp1 = bBase + (long)(n0 + 16 + (lane & 15)) * bStride + kSelB;
  const __bf16* bp2 = bBase + (long)(n0 + 32 + (lane & 15)) * bStride + kSelB;
  const __bf16* bp3 = bBase + (long)(n0 + 48 + (lane & 15)) * bStride + kSelB;

  v8f ac0 = {}, ac1 = {}, ac2 = {}, ac3 = {};
  for (int k = 0; k < K; k += 32) {
    v16bf a, b0, b1, b2, b3;
    *(v8bf*)&a        = *(const v8bf*)(aLds + k);
    *((v8bf*)&a + 1)  = *(const v8bf*)(aLds + k + 8);
    *(v8bf*)&b0       = *(const v8bf*)(bp0 + k);
    *((v8bf*)&b0 + 1) = *(const v8bf*)(bp0 + k + 8);
    *(v8bf*)&b1       = *(const v8bf*)(bp1 + k);
    *((v8bf*)&b1 + 1) = *(const v8bf*)(bp1 + k + 8);
    *(v8bf*)&b2       = *(const v8bf*)(bp2 + k);
    *((v8bf*)&b2 + 1) = *(const v8bf*)(bp2 + k + 8);
    *(v8bf*)&b3       = *(const v8bf*)(bp3 + k);
    *((v8bf*)&b3 + 1) = *(const v8bf*)(bp3 + k + 8);
    ac0 = wmma_bf16(a, b0, ac0);
    ac1 = wmma_bf16(a, b1, ac1);
    ac2 = wmma_bf16(a, b2, ac2);
    ac3 = wmma_bf16(a, b3, ac3);
  }

  const int col = lane & 15;
  const int rOff = (lane >> 4) << 3;
  const long rowBase = m0b + rt * 16 + rOff;
  float bv0 = 0.f, bv1 = 0.f, bv2 = 0.f, bv3 = 0.f;
  if constexpr (HASBIAS) {
    bv0 = bias[n0 + col];      bv1 = bias[n0 + 16 + col];
    bv2 = bias[n0 + 32 + col]; bv3 = bias[n0 + 48 + col];
  }
#pragma unroll
  for (int r = 0; r < 8; ++r) {
    float* cp = Cf + (rowBase + r) * Nc + n0 + col;
    cp[0]  = ac0[r] + bv0;
    cp[16] = ac1[r] + bv1;
    cp[32] = ac2[r] + bv2;
    cp[48] = ac3[r] + bv3;
  }
}

// ---------------------------------------------------------------------------
// Attention per batch n. Phase 1 transposes h into padded LDS (free: every
// element is read for the s1/s2 dots anyway). Phase 3: separable softmax ->
// A fragments regenerated on the fly, reused across 4 WMMA chains.
// Writes h' into cf[:, 0:256].
// ---------------------------------------------------------------------------
__global__ void attn_kernel(const float* __restrict__ h,     // N*V*H f32
                            const float* __restrict__ a_w,   // 2H
                            __bf16* __restrict__ cf) {       // M x 512
  __shared__ float s1[Vv], s2[Vv], mx[Vv], idn[Vv];
  extern __shared__ char smem[];
  __bf16* ldsHt = (__bf16*)smem;                 // [256 cols][264] bf16
  const int n = blockIdx.x;
  const int t = threadIdx.x;

  // phase 1: projections + transpose to LDS
  const float* hr = h + ((long)n * Vv + t) * Hh;
  float d1 = 0.f, d2 = 0.f;
  for (int k = 0; k < Hh; ++k) {
    float hv = hr[k];
    d1 += hv * a_w[k];
    d2 += hv * a_w[Hh + k];
    ldsHt[k * 264 + t] = f2bf(hv);
  }
  s1[t] = d1; s2[t] = d2;
  __syncthreads();

  // phase 2: row softmax stats over separable logits
  const float s1i = s1[t];
  float mloc = -3.4e38f;
  for (int j = 0; j < Vv; ++j) {
    float v = s1i + s2[j];
    v = v > 0.f ? v : 0.01f * v;
    mloc = v > mloc ? v : mloc;
  }
  float sum = 0.f;
  for (int j = 0; j < Vv; ++j) {
    float v = s1i + s2[j];
    v = v > 0.f ? v : 0.01f * v;
    sum += __expf(v - mloc);
  }
  mx[t] = mloc; idn[t] = 1.0f / sum;
  __syncthreads();

  // phase 3
  const int lane = t & 31, wave = t >> 5;
  const int kSelA = (lane >> 4) << 4;
  const int kSelB = (lane >> 4) << 4;
  for (int grp = wave; grp < 64; grp += 8) {     // 16 row-tiles x 4 col-groups
    const int m0 = (grp >> 2) << 4;
    const int n0 = (grp & 3) << 6;
    const int i  = m0 + (lane & 15);
    const float s1v = s1[i], mxv = mx[i], idv = idn[i];
    const __bf16* bp0 = ldsHt + (n0 + (lane & 15)) * 264 + kSelB;
    const __bf16* bp1 = bp0 + 16 * 264;
    const __bf16* bp2 = bp0 + 32 * 264;
    const __bf16* bp3 = bp0 + 48 * 264;

    v8f ac0 = {}, ac1 = {}, ac2 = {}, ac3 = {};
    for (int k0 = 0; k0 < Vv; k0 += 32) {
      v16bf a, b0, b1, b2, b3;
#pragma unroll
      for (int e = 0; e < 8; ++e) {
        int ka = k0 + kSelA + e;
        float v = s1v + s2[ka];
        v = v > 0.f ? v : 0.01f * v;
        a[e] = f2bf(__expf(v - mxv) * idv);
        float v2 = s1v + s2[ka + 8];
        v2 = v2 > 0.f ? v2 : 0.01f * v2;
        a[8 + e] = f2bf(__expf(v2 - mxv) * idv);
      }
      *(v8bf*)&b0       = *(const v8bf*)(bp0 + k0);
      *((v8bf*)&b0 + 1) = *(const v8bf*)(bp0 + k0 + 8);
      *(v8bf*)&b1       = *(const v8bf*)(bp1 + k0);
      *((v8bf*)&b1 + 1) = *(const v8bf*)(bp1 + k0 + 8);
      *(v8bf*)&b2       = *(const v8bf*)(bp2 + k0);
      *((v8bf*)&b2 + 1) = *(const v8bf*)(bp2 + k0 + 8);
      *(v8bf*)&b3       = *(const v8bf*)(bp3 + k0);
      *((v8bf*)&b3 + 1) = *(const v8bf*)(bp3 + k0 + 8);
      ac0 = wmma_bf16(a, b0, ac0);
      ac1 = wmma_bf16(a, b1, ac1);
      ac2 = wmma_bf16(a, b2, ac2);
      ac3 = wmma_bf16(a, b3, ac3);
    }
    const int rOff = (lane >> 4) << 3;
#pragma unroll
    for (int r = 0; r < 8; ++r) {
      __bf16* cp = cf + ((long)n * Vv + m0 + r + rOff) * CF_K + n0 + (lane & 15);
      cp[0]  = f2bf(ac0[r]);
      cp[16] = f2bf(ac1[r]);
      cp[32] = f2bf(ac2[r]);
      cp[48] = f2bf(ac3[r]);
    }
  }
}

// ---------------------------------------------------------------------------
// Batchnorm statistics / apply.
// ---------------------------------------------------------------------------
__global__ void zero_kernel(float* p, int n) {
  int i = blockIdx.x * 256 + threadIdx.x;
  if (i < n) p[i] = 0.f;
}

__global__ void bn_stats_kernel(const float* __restrict__ Y, int C,
                                float* __restrict__ sums, float* __restrict__ sqs) {
  int  c  = threadIdx.x;
  long r0 = (long)blockIdx.x * 1024;
  float s = 0.f, q = 0.f;
  for (long r = r0; r < r0 + 1024; ++r) {
    float v = Y[r * C + c];
    s += v; q += v * v;
  }
  atomicAdd(&sums[c], s);
  atomicAdd(&sqs[c],  q);
}

template <bool TOBF>
__global__ void bn_apply_kernel(const float* __restrict__ Y,
                                const float* __restrict__ sums,
                                const float* __restrict__ sqs,
                                const float* __restrict__ g,
                                const float* __restrict__ be,
                                int C, float invM,
                                __bf16* __restrict__ xb, float* __restrict__ xf) {
  long idx = (long)blockIdx.x * 256 + threadIdx.x;
  int  c   = (int)(idx % C);
  float mean = sums[c] * invM;
  float var  = sqs[c] * invM - mean * mean;
  float v = g[c] * (Y[idx] - mean) * rsqrtf(var + 1e-5f) + be[c];
  v = v > 0.f ? v : 0.f;
  if constexpr (TOBF) xb[idx] = f2bf(v);
  else                xf[idx] = v;
}

// ---------------------------------------------------------------------------
extern "C" void kernel_launch(void* const* d_in, const int* in_sizes, int n_in,
                              void* d_out, int out_size, void* d_ws, size_t ws_size,
                              hipStream_t stream) {
  (void)in_sizes; (void)n_in; (void)out_size; (void)ws_size;
  const float* obs = (const float*)d_in[0];
  const float* hi  = (const float*)d_in[1];
  const float* Ws  = (const float*)d_in[2];
  const float* bs  = (const float*)d_in[3];
  const float* Wv  = (const float*)d_in[4];
  const float* bv  = (const float*)d_in[5];
  const float* Wg  = (const float*)d_in[6];
  const float* a_w = (const float*)d_in[7];
  const float* W1  = (const float*)d_in[8];
  const float* b1  = (const float*)d_in[9];
  const float* g1  = (const float*)d_in[10];
  const float* be1 = (const float*)d_in[11];
  const float* W2  = (const float*)d_in[12];
  const float* b2  = (const float*)d_in[13];
  const float* g2  = (const float*)d_in[14];
  const float* be2 = (const float*)d_in[15];
  float* out = (float*)d_out;

  char*  ws  = (char*)d_ws;
  size_t off = 0;
  auto alloc = [&](size_t bytes) {
    char* p = ws + off;
    off += (bytes + 255) & ~(size_t)255;
    return p;
  };
  __bf16* cf    = (__bf16*)alloc((size_t)Mrows * CF_K * 2);   // live to gemm2
  char*   regA  = alloc((size_t)Mrows * GIN * 2);             // comb -> y1
  char*   regB  = alloc((size_t)Mrows * Hh * 4);              // h    -> x1
  char*   regC  = alloc((size_t)Mrows * OUTC * 4);            // y2
  __bf16* Wgt   = (__bf16*)alloc((size_t)Hh * GIN * 2);
  __bf16* W1t   = (__bf16*)alloc((size_t)Hh * CF_K * 2);
  __bf16* W2t   = (__bf16*)alloc((size_t)OUTC * Hh * 2);
  float*  stats = (float*)alloc(768 * 4);

  __bf16* comb = (__bf16*)regA;  float*  y1 = (float*)regA;
  float*  hbuf = (float*)regB;   __bf16* x1 = (__bf16*)regB;
  float*  y2   = (float*)regC;
  float *sum1 = stats, *sq1 = stats + 256, *sum2 = stats + 512, *sq2 = stats + 640;

  // 1) weight prep
  wprep_kernel<<<(GIN * Hh + 255) / 256, 256, 0, stream>>>(Wg, Wgt, GIN, Hh);
  wprep_kernel<<<(CF_K * Hh + 255) / 256, 256, 0, stream>>>(W1, W1t, CF_K, Hh);
  wprep_kernel<<<(Hh * OUTC + 255) / 256, 256, 0, stream>>>(W2, W2t, Hh, OUTC);

  // 2) embed
  embed_kernel<<<Mrows, 256, 0, stream>>>(obs, hi, Ws, bs, Wv, bv, comb, cf);

  // 3) h = comb @ Wg   (RB=32 rows/block, LDS A = 32*(576+8)*2 B)
  gemm_bf16_kernel<false, false><<<Mrows / 32, 256, 32 * (GIN + 8) * 2, stream>>>(
      comb, Wgt, nullptr, hbuf, GIN, Hh);

  // 4) attention -> cf[:,0:256]   (dynamic LDS: 256*264 bf16)
  attn_kernel<<<Nn, 256, 256 * 264 * 2, stream>>>(hbuf, a_w, cf);

  // 5) y1 = cf @ W1 + b1
  gemm_bf16_kernel<true, false><<<Mrows / 32, 256, 32 * (CF_K + 8) * 2, stream>>>(
      cf, W1t, b1, y1, CF_K, Hh);

  // 6) batchnorm-1 -> bf16 x1
  zero_kernel<<<3, 256, 0, stream>>>(stats, 768);
  bn_stats_kernel<<<Mrows / 1024, Hh, 0, stream>>>(y1, Hh, sum1, sq1);
  bn_apply_kernel<true><<<(size_t)Mrows * Hh / 256, 256, 0, stream>>>(
      y1, sum1, sq1, g1, be1, Hh, 1.0f / Mrows, x1, nullptr);

  // 7) y2 = x1 @ W2 + b2   (RB=64; LDS = A 64*264*2 + B 128*264*2; B via TDM)
  gemm_bf16_kernel<true, true><<<Mrows / 64, 256,
                                 64 * (Hh + 8) * 2 + 128 * 264 * 2, stream>>>(
      x1, W2t, b2, y2, Hh, OUTC);

  // 8) batchnorm-2 -> f32 out
  bn_stats_kernel<<<Mrows / 1024, OUTC, 0, stream>>>(y2, OUTC, sum2, sq2);
  bn_apply_kernel<false><<<(size_t)Mrows * OUTC / 256, 256, 0, stream>>>(
      y2, sum2, sq2, g2, be2, OUTC, 1.0f / Mrows, nullptr, out);
}
